// NN_Model_70007966925177
// MI455X (gfx1250) — matrix operand order repaired and checked by reading, not compiled, not run on hardware
//
#include <hip/hip_runtime.h>
#include <hip/hip_fp16.h>

// ---------------- problem constants ----------------
#define DH      256                 // D_H
#define DD1     785                 // DATA_DIM + 1
#define LEN_M   (DD1 * DH)          // 200960
#define LEN_PRI (DH + 1)            // 257
#define NMC     64                  // NUM_MC
#define NB      2048                // N_BATCH
#define KPAD    800                 // DD1 padded up to 25 * 32 for K-loop
#define BKP     40                  // LDS transpose-tile k stride (f16), padded

typedef __attribute__((ext_vector_type(16))) _Float16 v16h;
typedef __attribute__((ext_vector_type(8)))  float    v8f;

union Frag16 { v16h v; uint4 u[2]; };

__device__ __forceinline__ float softplus_f(float v) {
    // stable logaddexp(v, 0)
    return fmaxf(v, 0.f) + log1pf(expf(-fabsf(v)));
}

// ---------------- prep: x (f32, 2048x785) -> x16 (f16, 2048x800, zero padded) ----
__global__ void prep_x_kernel(const float* __restrict__ x, _Float16* __restrict__ x16) {
    int i = blockIdx.x * blockDim.x + threadIdx.x;
    if (i >= NB * KPAD) return;
    int r = i / KPAD, c = i - r * KPAD;
    float v = (c < DD1) ? x[r * DD1 + c] : 0.f;
    x16[i] = (_Float16)v;
}

// ---------------- prep: sv[i] = sqrt(softplus(raw_v_w0[i]) + 1e-6); partial sums --
__global__ void prep_sv_kernel(const float* __restrict__ params,
                               float* __restrict__ sv,
                               float* __restrict__ partials) {
    __shared__ float s1[256], s2[256];
    const int t = threadIdx.x;
    float a1 = 0.f, a2 = 0.f;
    for (int i = blockIdx.x * blockDim.x + t; i < LEN_M; i += gridDim.x * blockDim.x) {
        float raw = params[LEN_M + LEN_PRI + i];
        float s = sqrtf(softplus_f(raw) + 1e-6f);
        sv[i] = s;
        a1 += s;                       // sum(sqrt_v_w0)
        float m = params[i];
        a2 += m * m;                   // sum(m_w0^2)
    }
    s1[t] = a1; s2[t] = a2;
    __syncthreads();
    for (int off = 128; off > 0; off >>= 1) {
        if (t < off) { s1[t] += s1[t + off]; s2[t] += s2[t + off]; }
        __syncthreads();
    }
    if (t == 0) { partials[blockIdx.x] = s1[0]; partials[256 + blockIdx.x] = s2[0]; }
}

// ---------------- small outputs: m_pri, v_pri, KL scalar -------------------------
__global__ void finalize_kernel(const float* __restrict__ params,
                                const float* __restrict__ partials,
                                float* __restrict__ out) {
    __shared__ float s3[256], s4[256];
    const int t = threadIdx.x;
    float a3 = 0.f, a4 = 0.f;
    for (int i = t; i < LEN_PRI; i += 256) {
        float mp   = params[LEN_M + i];
        float vraw = params[LEN_M + LEN_PRI + LEN_M + i];
        float vp   = softplus_f(vraw);
        out[i] = mp;                   // m_pri
        out[LEN_PRI + i] = vp;         // v_pri
        a3 += sqrtf(vp + 1e-6f);       // sum(sqrt_v_pri)
        a4 += mp * mp;                 // sum(m_pri^2)
    }
    s3[t] = a3; s4[t] = a4;
    __syncthreads();
    for (int off = 128; off > 0; off >>= 1) {
        if (t < off) { s3[t] += s3[t + off]; s4[t] += s4[t + off]; }
        __syncthreads();
    }
    if (t == 0) {
        float S1 = 0.f, S2 = 0.f;
        for (int b = 0; b < 256; ++b) { S1 += partials[b]; S2 += partials[256 + b]; }
        float S3 = s3[0], S4 = s4[0];
        // LAMB == 1 -> log(lamb) terms vanish
        float KL_w1 = 0.5f * (S3 - (float)LEN_PRI + S4 - logf(S3));
        float KL_w0 = 0.5f * (S1 - (float)DD1     + S2 - logf(S1));
        out[2 * (size_t)LEN_PRI + (size_t)NB * DH * NMC] = KL_w1 + KL_w0;
    }
}

// ---------------- prep: synthesize W16t[h][m][d] (f16, d padded to KPAD) ---------
// W[d,h,m] = m_w0[d*256+h] + eps[(d*256+h)*64+m] * sv[d*256+h]; transposed so the
// GEMM's B fragments are contiguous-in-K global loads. One block = (32 d) x (64 m)
// for one h; contiguous 32B eps reads, LDS transpose, 16B coalesced f16 writes.
__global__ __launch_bounds__(256)
void prep_w_kernel(const float* __restrict__ params,
                   const float* __restrict__ sv,
                   const float* __restrict__ eps,
                   _Float16* __restrict__ w16) {
    __shared__ __align__(16) _Float16 T[64 * BKP];   // [m][k]
    const int t  = threadIdx.x;
    const int h  = blockIdx.y;
    const int d0 = blockIdx.x * 32;
    const int k  = t & 31;
    const int mg = (t >> 5) << 3;
    const int d  = d0 + k;

    float w[8];
    if (d < DD1) {
        const int idx  = d * DH + h;
        const float mw = params[idx];
        const float s  = sv[idx];
        const float4* e = (const float4*)(eps + (size_t)idx * NMC + mg);
        const float4 e0 = e[0], e1 = e[1];
        w[0] = fmaf(e0.x, s, mw); w[1] = fmaf(e0.y, s, mw);
        w[2] = fmaf(e0.z, s, mw); w[3] = fmaf(e0.w, s, mw);
        w[4] = fmaf(e1.x, s, mw); w[5] = fmaf(e1.y, s, mw);
        w[6] = fmaf(e1.z, s, mw); w[7] = fmaf(e1.w, s, mw);
    } else {
#pragma unroll
        for (int i = 0; i < 8; ++i) w[i] = 0.f;
    }
#pragma unroll
    for (int i = 0; i < 8; ++i)
        T[(mg + i) * BKP + k] = (_Float16)w[i];
    __syncthreads();

    // write 64 m-rows x 32 k f16: thread -> (row = t/4, 8-f16 chunk = t%4)
    const int row = t >> 2;
    const int ch  = (t & 3) * 8;
    const uint4 v = *(const uint4*)(T + row * BKP + ch);
    *(uint4*)(w16 + ((size_t)h * NMC + row) * KPAD + d0 + ch) = v;
}

// ---------------- main GEMM: relu(x @ W)[n,h,m], WMMA f16 -> f32 -----------------
// No LDS, no barriers: A from x16 (row-major, KPAD stride), B from W16t
// (K-contiguous). 8 waves = 4 n-groups x 2 m-groups; each wave: 64n x 32m via
// 4x2 fragments (8 WMMAs per 32-K step).
__global__ __launch_bounds__(256)
void gemm_relu_kernel(const _Float16* __restrict__ x16,
                      const _Float16* __restrict__ w16,
                      float*          __restrict__ out) {
    const int tid  = threadIdx.x;
    const int lane = tid & 31;
    const int wid  = tid >> 5;
    const int wn   = wid & 3;          // 4 n groups, 64 rows each
    const int wm   = wid >> 2;         // 2 m groups, 32 cols each
    const int half = lane >> 4;
    const int l16  = lane & 15;

    const int h  = blockIdx.y;
    const int n0 = blockIdx.x * 256 + wn * 64;
    const int m0 = wm * 32;

    v8f c[4][2] = {};

    // A: lane = M row; chunks of 8 K at half*8 and 16+half*8
    const _Float16* pa[4];
#pragma unroll
    for (int i = 0; i < 4; ++i)
        pa[i] = x16 + (size_t)(n0 + i * 16 + l16) * KPAD + half * 8;
    // B: lane = N col; chunks of 8 K at half*8 and 16+half*8
    const _Float16* pb[2];
#pragma unroll
    for (int j = 0; j < 2; ++j)
        pb[j] = w16 + ((size_t)h * NMC + m0 + j * 16 + l16) * KPAD + half * 8;

    for (int k0 = 0; k0 < KPAD; k0 += 32) {
        Frag16 a[4], b[2];
#pragma unroll
        for (int i = 0; i < 4; ++i) {
            a[i].u[0] = *(const uint4*)(pa[i] + k0);
            a[i].u[1] = *(const uint4*)(pa[i] + k0 + 16);
        }
#pragma unroll
        for (int j = 0; j < 2; ++j) {
            b[j].u[0] = *(const uint4*)(pb[j] + k0);
            b[j].u[1] = *(const uint4*)(pb[j] + k0 + 16);
        }
        // speculative prefetch of next K-step of B (OOB-safe, dropped silently)
        __builtin_prefetch((const void*)(pb[0] + k0 + 32), 0, 1);

#pragma unroll
        for (int i = 0; i < 4; ++i)
#pragma unroll
            for (int j = 0; j < 2; ++j)
                c[i][j] = __builtin_amdgcn_wmma_f32_16x16x32_f16(
                    false, a[i].v, false, b[j].v, (short)0, c[i][j], false, false);
    }

    // epilogue: relu + store (lane = m -> coalesced 64B per half-wave)
    float* outp = out + 2 * LEN_PRI;
#pragma unroll
    for (int i = 0; i < 4; ++i)
#pragma unroll
        for (int j = 0; j < 2; ++j)
#pragma unroll
            for (int r = 0; r < 8; ++r) {
                const int n = n0 + i * 16 + r + 8 * half;
                const int m = m0 + j * 16 + l16;
                float v = c[i][j][r];
                outp[(size_t)n * (DH * NMC) + (size_t)h * NMC + m] = v > 0.f ? v : 0.f;
            }
}

// ---------------- launcher ----------------
extern "C" void kernel_launch(void* const* d_in, const int* in_sizes, int n_in,
                              void* d_out, int out_size, void* d_ws, size_t ws_size,
                              hipStream_t stream) {
    const float* x      = (const float*)d_in[0];   // (2048, 785)
    const float* params = (const float*)d_in[1];   // (PARAM_LEN,)
    const float* eps    = (const float*)d_in[2];   // (200960, 64)
    float* out = (float*)d_out;

    char* ws = (char*)d_ws;
    size_t off = 0;
    _Float16* x16   = (_Float16*)(ws + off); off += (size_t)NB * KPAD * sizeof(_Float16); // 3,276,800
    float* sv       = (float*)(ws + off);    off += (size_t)LEN_M * sizeof(float);        //   803,840
    float* partials = (float*)(ws + off);    off += 512 * sizeof(float);                  //     2,048
    _Float16* w16   = (_Float16*)(ws + off); // 256*64*800*2 = 26,214,400 B  (total ~30.3 MB)

    prep_x_kernel<<<(NB * KPAD + 255) / 256, 256, 0, stream>>>(x, x16);
    prep_sv_kernel<<<256, 256, 0, stream>>>(params, sv, partials);
    finalize_kernel<<<1, 256, 0, stream>>>(params, partials, out);

    dim3 wgrid(KPAD / 32, DH);    // (25, 256)
    prep_w_kernel<<<wgrid, 256, 0, stream>>>(params, sv, eps, w16);

    dim3 grid(NB / 256, DH);      // (8, 256)
    gemm_relu_kernel<<<grid, 256, 0, stream>>>(x16, w16, out);
}
